// OCRHead_48026324304272
// MI455X (gfx1250) — matrix operand order repaired and checked by reading, not compiled, not run on hardware
//
#include <hip/hip_runtime.h>
#include <hip/hip_bf16.h>
#include <stdint.h>

#define BATCH 4
#define CIN   720
#define CINP  736          // cin padded to multiple of 32
#define NCH   23           // CINP/32 conv K-chunks
#define HH    128
#define WW    128
#define PIX   (HH*WW)      // 16384
#define MIDC  512
#define NCLS  19
#define KCH   256
#define VCH   256
#define NTAPS 9
#define NT_CONV (MIDC/16)  // 32

typedef __attribute__((ext_vector_type(16))) __bf16       v16bf;
typedef __attribute__((ext_vector_type(8)))  float        v8f;
typedef __attribute__((ext_vector_type(4)))  unsigned int u32x4;

union FragAB { v16bf v; u32x4 q[2]; };

#define WMMA_BF16(A_, B_, C_) \
    __builtin_amdgcn_wmma_f32_16x16x32_bf16(false, (A_), false, (B_), (short)0, (C_), false, false)

__device__ __forceinline__ unsigned short f2bf(float f) {
    union { float f; unsigned u; } x; x.f = f;
    unsigned u = x.u;
    unsigned r = u + 0x7fffu + ((u >> 16) & 1u);   // round-to-nearest-even
    return (unsigned short)(r >> 16);
}

// ---- feats f32 [b][720][P] -> bf16 interleaved [b][23][P][32] --------------
__global__ void k_cvt_feats(const float* __restrict__ feats,
                            unsigned short* __restrict__ fbT, long total) {
    long i = (long)blockIdx.x * blockDim.x + threadIdx.x;
    if (i >= total) return;
    int kk = (int)(i & 31);
    int p  = (int)((i >> 5) % PIX);
    int ch = (int)((i >> 5) / PIX % NCH);
    int b  = (int)(i / ((long)32 * PIX * NCH));
    int c  = ch * 32 + kk;
    float v = 0.f;
    if (c < CIN) v = feats[((long)(b * CIN + c)) * PIX + p];
    fbT[i] = f2bf(v);
}

// ---- conv weights -> WMMA-B packed, stream-ordered [tap][ntile][chunk] -----
__global__ void k_pack_conv(const float* __restrict__ w3,
                            unsigned int* __restrict__ dst, int total) {
    int i = blockIdx.x * blockDim.x + threadIdx.x;
    if (i >= total) return;
    int d     = i & 7;
    int lane  = (i >> 3) & 31;
    int chunk = (i >> 8) % NCH;
    int ntile = (i / (256 * NCH)) & 31;
    int tap   = i / (256 * NCH * 32);
    int mid = ntile * 16 + (lane & 15);
    int kk  = chunk * 32 + (d >> 2) * 16 + (lane >> 4) * 8 + (d & 3) * 2;
    int ky = tap / 3, kx = tap % 3;
    unsigned short lo = 0, hi = 0;
    if (kk     < CIN) lo = f2bf(w3[(((long)mid * CIN + kk    ) * 3 + ky) * 3 + kx]);
    if (kk + 1 < CIN) hi = f2bf(w3[(((long)mid * CIN + kk + 1) * 3 + ky) * 3 + kx]);
    dst[i] = ((unsigned)hi << 16) | lo;
}

// ---- generic [nOut x kIn] f32 weight -> WMMA-B packed [ntile][chunk] -------
__global__ void k_pack_mat(const float* __restrict__ W, unsigned int* __restrict__ dst,
                           int nOut, int kIn, int NT, int chunks, int transposed) {
    int total = chunks * NT * 256;
    for (int i = blockIdx.x * blockDim.x + threadIdx.x; i < total;
         i += gridDim.x * blockDim.x) {
        int d     = i & 7;
        int lane  = (i >> 3) & 31;
        int chunk = (i >> 8) % chunks;
        int ntile = i / (256 * chunks);
        int n  = ntile * 16 + (lane & 15);
        int kk = chunk * 32 + (d >> 2) * 16 + (lane >> 4) * 8 + (d & 3) * 2;
        float lo = 0.f, hi = 0.f;
        if (n < nOut) {
            if (kk     < kIn) lo = transposed ? W[(long)kk * nOut + n]       : W[(long)n * kIn + kk];
            if (kk + 1 < kIn) hi = transposed ? W[(long)(kk + 1) * nOut + n] : W[(long)n * kIn + kk + 1];
        }
        dst[i] = ((unsigned)f2bf(hi) << 16) | f2bf(lo);
    }
}

// ---- conv3x3 implicit GEMM: 2x2 register blocking, halo-slab staging -------
// grid (PIX/32, 2, BATCH), block 256. WG = 32 px x 256 mid; wave = 32px x 32mid.
__global__ void __launch_bounds__(256)
k_conv3x3(const unsigned short* __restrict__ fbT,   // [b][23][P][32]
          const unsigned int* __restrict__ Bp,
          float* __restrict__ out) {
    __shared__ __align__(16) unsigned short sX[NCH * 34 * 40];  // 61 KB halo slab
    int tid  = threadIdx.x;
    int lane = tid & 31;
    int wave = tid >> 5;
    int p0  = blockIdx.x * 32;
    int ntp = blockIdx.y * 8 + wave;      // ntile-pair: mids [ntp*32, ntp*32+32)
    int b   = blockIdx.z;
    int y  = p0 >> 7;
    int x0 = p0 & 127;
    v8f acc00 = {}, acc01 = {}, acc10 = {}, acc11 = {};
    const u32x4* bq = (const u32x4*)Bp;
    int m = lane & 15, half = lane >> 4;
    for (int ry = 0; ry < 3; ++ry) {
        int ys = y + ry - 1;
        __syncthreads();
        // stage 23 chunks x 34 halo pixels, 128b granules, coalesced
        for (int idx = tid; idx < NCH * 34 * 4; idx += 256) {
            int seg = idx & 3;
            int xi  = (idx >> 2) % 34;
            int ch  = idx / 136;
            int xs  = x0 - 1 + xi;
            u32x4 val = {};
            if (ys >= 0 && ys < HH && xs >= 0 && xs < WW) {
                long g = ((long)(b * NCH + ch) * PIX + ys * WW + xs) * 32 + seg * 8;
                val = *(const u32x4*)(fbT + g);
            }
            *(u32x4*)&sX[(ch * 34 + xi) * 40 + seg * 8] = val;
        }
        __syncthreads();
        for (int dx = 0; dx < 3; ++dx) {
            int t = ry * 3 + dx;
            int wbase = ((t * NT_CONV + ntp * 2) * NCH) * 64;   // u32x4 units
            for (int ch = 0; ch < NCH; ++ch) {
                FragAB a0, a1, w0, w1;
                int ab = (ch * 34 + m + dx) * 40 + half * 8;
                a0.q[0] = *(const u32x4*)&sX[ab];
                a0.q[1] = *(const u32x4*)&sX[ab + 16];
                a1.q[0] = *(const u32x4*)&sX[ab + 16 * 40];
                a1.q[1] = *(const u32x4*)&sX[ab + 16 * 40 + 16];
                int i4 = wbase + ch * 64 + lane * 2;
                w0.q[0] = bq[i4];
                w0.q[1] = bq[i4 + 1];
                w1.q[0] = bq[i4 + NCH * 64];
                w1.q[1] = bq[i4 + NCH * 64 + 1];
                acc00 = WMMA_BF16(a0.v, w0.v, acc00);
                acc01 = WMMA_BF16(a0.v, w1.v, acc01);
                acc10 = WMMA_BF16(a1.v, w0.v, acc10);
                acc11 = WMMA_BF16(a1.v, w1.v, acc11);
            }
        }
    }
    int n  = lane & 15;
    int hl = lane >> 4;
    long c0 = (long)(b * MIDC + ntp * 32 + n) * PIX;
    long c1 = c0 + (long)16 * PIX;
#pragma unroll
    for (int i = 0; i < 8; ++i) {
        int pp = p0 + i + 8 * hl;
        out[c0 + pp]      = acc00[i];
        out[c1 + pp]      = acc01[i];
        out[c0 + pp + 16] = acc10[i];
        out[c1 + pp + 16] = acc11[i];
    }
}

// ---- epilogue helper for 1x1 GEMM ------------------------------------------
__device__ __forceinline__ void gemm_epi(v8f acc, int b, int n, int pbase, int hl,
        float scale, const float* __restrict__ bias, const float* __restrict__ addx,
        int nValid, int outF32, float* __restrict__ outF,
        unsigned short* __restrict__ outB, int outChunks) {
#pragma unroll
    for (int i = 0; i < 8; ++i) {
        int p = pbase + i + 8 * hl;
        float v = acc[i] * scale;
        if (bias && n < nValid) v += bias[n];
        if (addx) v += addx[((long)(b * MIDC) + n) * PIX + p];
        if (outF32) {
            if (n < nValid) outF[((long)(b * nValid) + n) * PIX + p] = v;
        } else {
            outB[(((long)(b * outChunks) + (n >> 5)) * PIX + p) * 32 + (n & 31)] = f2bf(v);
        }
    }
}

// ---- generic 1x1 GEMM: A interleaved [b][chunks][P][32], 2x2 blocking ------
// grid (PIX/32, ceil((NT/2)/8), BATCH), block 32*min(NT/2,8)
__global__ void __launch_bounds__(256)
k_gemm1x1(const unsigned short* __restrict__ A,
          const unsigned int* __restrict__ Bp, long bStrideDw,
          int chunks, int NT,
          float scale, const float* __restrict__ bias,
          const float* __restrict__ addx,         // optional f32 [b][MIDC][P]
          int nValid, int outF32,
          float* __restrict__ outF, unsigned short* __restrict__ outB,
          int outChunks) {
    __shared__ __align__(16) unsigned short sA[16 * 32 * 40];   // up to K=512, 32 px
    int tid  = threadIdx.x;
    int lane = tid & 31;
    int wave = tid >> 5;
    int p0  = blockIdx.x * 32;
    int ntp = blockIdx.y * 8 + wave;     // pair id: n in [ntp*32, ntp*32+32)
    int b   = blockIdx.z;
    int npairs = NT >> 1;
    bool active = (ntp < npairs);
    v8f acc00 = {}, acc01 = {}, acc10 = {}, acc11 = {};
    const u32x4* bq = (const u32x4*)(Bp + (long)b * bStrideDw);
    for (int idx = tid; idx < chunks * 128; idx += blockDim.x) {
        int seg = idx & 3;
        int mm  = (idx >> 2) & 31;
        int ch  = idx >> 7;
        long g = ((long)(b * chunks + ch) * PIX + p0 + mm) * 32 + seg * 8;
        *(u32x4*)&sA[(ch * 32 + mm) * 40 + seg * 8] = *(const u32x4*)(A + g);
    }
    __syncthreads();
    if (active) {
        int m = lane & 15, half = lane >> 4;
        int wbase = (ntp * 2) * chunks * 64;
        for (int ch = 0; ch < chunks; ++ch) {
            FragAB a0, a1, w0, w1;
            int ab = (ch * 32 + m) * 40 + half * 8;
            a0.q[0] = *(const u32x4*)&sA[ab];
            a0.q[1] = *(const u32x4*)&sA[ab + 16];
            a1.q[0] = *(const u32x4*)&sA[ab + 16 * 40];
            a1.q[1] = *(const u32x4*)&sA[ab + 16 * 40 + 16];
            int i4 = wbase + ch * 64 + lane * 2;
            w0.q[0] = bq[i4];
            w0.q[1] = bq[i4 + 1];
            w1.q[0] = bq[i4 + chunks * 64];
            w1.q[1] = bq[i4 + chunks * 64 + 1];
            acc00 = WMMA_BF16(a0.v, w0.v, acc00);
            acc01 = WMMA_BF16(a0.v, w1.v, acc01);
            acc10 = WMMA_BF16(a1.v, w0.v, acc10);
            acc11 = WMMA_BF16(a1.v, w1.v, acc11);
        }
    }
    if (!active) return;
    int nlo = ntp * 32 + (lane & 15);
    int hl  = lane >> 4;
    gemm_epi(acc00, b, nlo,      p0,      hl, scale, bias, addx, nValid, outF32, outF, outB, outChunks);
    gemm_epi(acc01, b, nlo + 16, p0,      hl, scale, bias, addx, nValid, outF32, outF, outB, outChunks);
    gemm_epi(acc10, b, nlo,      p0 + 16, hl, scale, bias, addx, nValid, outF32, outF, outB, outChunks);
    gemm_epi(acc11, b, nlo + 16, p0 + 16, hl, scale, bias, addx, nValid, outF32, outF, outB, outChunks);
}

// ---- BatchNorm stats (per channel over B*H*W) ------------------------------
__global__ void __launch_bounds__(256)
k_bnstats(const float* __restrict__ x, const float* __restrict__ gamma,
          const float* __restrict__ beta, float* __restrict__ sa, float* __restrict__ sb) {
    __shared__ float rs[256], rq[256];
    int c = blockIdx.x;
    float s = 0.f, q = 0.f;
    for (int b = 0; b < BATCH; ++b) {
        const float* base = x + ((long)(b * MIDC + c)) * PIX;
        for (int p = threadIdx.x; p < PIX; p += 256) {
            float v = base[p];
            s += v; q += v * v;
        }
    }
    rs[threadIdx.x] = s; rq[threadIdx.x] = q;
    __syncthreads();
    for (int off = 128; off > 0; off >>= 1) {
        if (threadIdx.x < off) {
            rs[threadIdx.x] += rs[threadIdx.x + off];
            rq[threadIdx.x] += rq[threadIdx.x + off];
        }
        __syncthreads();
    }
    if (threadIdx.x == 0) {
        float N = (float)(BATCH * PIX);
        float mean = rs[0] / N;
        float var  = rq[0] / N - mean * mean;
        float a = gamma[c] * rsqrtf(var + 1e-5f);
        sa[c] = a;
        sb[c] = beta[c] - mean * a;
    }
}

// ---- BN affine + relu: f32 in place, bf16 interleaved [b][16][P][32] -------
__global__ void k_bnapply(float* __restrict__ x, unsigned short* __restrict__ xb,
                          const float* __restrict__ sa, const float* __restrict__ sb) {
    long i = (long)blockIdx.x * 256 + threadIdx.x;
    int p = (int)(i % PIX);
    int c = (int)((i / PIX) % MIDC);
    int b = (int)(i / ((long)PIX * MIDC));
    float v = x[i] * sa[c] + sb[c];
    v = v > 0.f ? v : 0.f;
    x[i] = v;
    xb[(((long)(b * 16) + (c >> 5)) * PIX + p) * 32 + (c & 31)] = f2bf(v);
}

__global__ void k_zero(float* p, int n) {
    int i = blockIdx.x * 256 + threadIdx.x;
    if (i < n) p[i] = 0.f;
}

// ---- spatial gather: per-pixel softmax + per-class denom -------------------
__global__ void __launch_bounds__(256)
k_softmax_gather(const float* __restrict__ aux, float* __restrict__ probs,
                 float* __restrict__ denom) {
    __shared__ float sden[NCLS];
    int b = blockIdx.y;
    int p = blockIdx.x * 256 + threadIdx.x;
    if (threadIdx.x < NCLS) sden[threadIdx.x] = 0.f;
    __syncthreads();
    float l[NCLS];
    float mx = -1e30f;
    for (int k = 0; k < NCLS; ++k) {
        l[k] = aux[((long)(b * NCLS + k)) * PIX + p];
        mx = fmaxf(mx, l[k]);
    }
    float s = 0.f;
    for (int k = 0; k < NCLS; ++k) { l[k] = __expf(l[k] - mx); s += l[k]; }
    float inv = 1.f / s;
    for (int k = 0; k < NCLS; ++k) {
        float pr = l[k] * inv;
        probs[((long)(b * NCLS + k)) * PIX + p] = pr;
        atomicAdd(&sden[k], pr);
    }
    __syncthreads();
    if (threadIdx.x < NCLS) atomicAdd(&denom[b * NCLS + threadIdx.x], sden[threadIdx.x]);
}

// ---- context[b][k][c] = sum_p w[p,k] * x[c,p] ------------------------------
__global__ void __launch_bounds__(256)
k_context(const float* __restrict__ probs, const float* __restrict__ denom,
          const float* __restrict__ x, float* __restrict__ ctx) {
    __shared__ float red[256];
    int c = blockIdx.x, k = blockIdx.y, b = blockIdx.z;
    const float* pr = probs + ((long)(b * NCLS + k)) * PIX;
    const float* xv = x + ((long)(b * MIDC + c)) * PIX;
    float s = 0.f;
    for (int p = threadIdx.x; p < PIX; p += 256) s += pr[p] * xv[p];
    red[threadIdx.x] = s;
    __syncthreads();
    for (int off = 128; off > 0; off >>= 1) {
        if (threadIdx.x < off) red[threadIdx.x] += red[threadIdx.x + off];
        __syncthreads();
    }
    if (threadIdx.x == 0) {
        float d = denom[b * NCLS + k];
        d = d < 1e-6f ? 1e-6f : d;
        ctx[((long)(b * NCLS) + k) * MIDC + c] = red[0] / d;
    }
}

// ---- k/v = context @ {k_w,v_w}^T + bias ------------------------------------
__global__ void __launch_bounds__(256)
k_kv(const float* __restrict__ ctx,
     const float* __restrict__ kw, const float* __restrict__ kb,
     const float* __restrict__ vw, const float* __restrict__ vb,
     float* __restrict__ kmat, float* __restrict__ vmat) {
    __shared__ float sc[MIDC];
    int k = blockIdx.x, b = blockIdx.y;
    const float* row = ctx + ((long)(b * NCLS) + k) * MIDC;
    for (int c = threadIdx.x; c < MIDC; c += 256) sc[c] = row[c];
    __syncthreads();
    int d = threadIdx.x;
    float aK = kb[d], aV = vb[d];
    for (int c = 0; c < MIDC; ++c) {
        float cv = sc[c];
        aK += cv * kw[(long)d * MIDC + c];
        aV += cv * vw[(long)d * MIDC + c];
    }
    kmat[((long)(b * NCLS) + k) * KCH + d] = aK;
    vmat[((long)(b * NCLS) + k) * VCH + d] = aV;
}

// ---- attention softmax over classes -> bf16 interleaved [b][1][P][32] ------
__global__ void k_softmax_attn(const float* __restrict__ scores,
                               unsigned short* __restrict__ attnb) {
    int b = blockIdx.y;
    int p = blockIdx.x * 256 + threadIdx.x;
    float l[NCLS];
    float mx = -1e30f;
    for (int k = 0; k < NCLS; ++k) {
        l[k] = scores[((long)(b * NCLS + k)) * PIX + p];
        mx = fmaxf(mx, l[k]);
    }
    float s = 0.f;
    for (int k = 0; k < NCLS; ++k) { l[k] = __expf(l[k] - mx); s += l[k]; }
    float inv = 1.f / s;
    long base = ((long)b * PIX + p) * 32;
    for (int k = 0; k < NCLS; ++k) attnb[base + k] = f2bf(l[k] * inv);
    for (int k = NCLS; k < 32; ++k) attnb[base + k] = 0;
}

// ============================================================================
extern "C" void kernel_launch(void* const* d_in, const int* in_sizes, int n_in,
                              void* d_out, int out_size, void* d_ws, size_t ws_size,
                              hipStream_t stream) {
    const float* feats = (const float*)d_in[0];
    const float* w3    = (const float*)d_in[1];
    const float* gamma = (const float*)d_in[2];
    const float* beta  = (const float*)d_in[3];
    const float* aux_w = (const float*)d_in[4];
    const float* aux_b = (const float*)d_in[5];
    const float* q_w   = (const float*)d_in[6];
    const float* k_w   = (const float*)d_in[7];
    const float* k_b   = (const float*)d_in[8];
    const float* v_w   = (const float*)d_in[9];
    const float* v_b   = (const float*)d_in[10];
    const float* out_w = (const float*)d_in[11];
    const float* cls_w = (const float*)d_in[12];
    const float* cls_b = (const float*)d_in[13];

    float* logits = (float*)d_out;                         // [4][19][P]
    float* auxout = logits + (long)BATCH * NCLS * PIX;     // [4][19][P]

    char* w = (char*)d_ws;
    auto alloc = [&](size_t bytes) {
        char* p = w;
        w += (bytes + 255) & ~(size_t)255;
        return p;
    };
    unsigned int*   convW   = (unsigned int*)alloc((size_t)NTAPS * 32 * NCH * 256 * 4);
    float*          xbuf    = (float*)alloc((size_t)BATCH * MIDC * PIX * 4);
    unsigned short* xb      = (unsigned short*)alloc((size_t)BATCH * MIDC * PIX * 2);
    float*          sa      = (float*)alloc(MIDC * 4);
    float*          sb      = (float*)alloc(MIDC * 4);
    unsigned int*   auxpack = (unsigned int*)alloc(16 * 2 * 256 * 4);
    unsigned int*   qpack   = (unsigned int*)alloc(16 * 16 * 256 * 4);
    unsigned int*   clspack = (unsigned int*)alloc(16 * 2 * 256 * 4);
    unsigned int*   owpack  = (unsigned int*)alloc(8 * 32 * 256 * 4);
    float*          ctx     = (float*)alloc((size_t)BATCH * NCLS * MIDC * 4);
    float*          denom   = (float*)alloc(BATCH * NCLS * 4);
    float*          kmat    = (float*)alloc((size_t)BATCH * NCLS * KCH * 4);
    float*          vmat    = (float*)alloc((size_t)BATCH * NCLS * VCH * 4);
    unsigned int*   kpack   = (unsigned int*)alloc((size_t)BATCH * 4096 * 4);
    unsigned int*   vpack   = (unsigned int*)alloc((size_t)BATCH * 4096 * 4);
    // big region: bf16 interleaved feats during conv, recycled afterwards
    char* big = alloc((size_t)BATCH * NCH * PIX * 32 * 2);     // 96.5 MB
    unsigned short* fbT    = (unsigned short*)big;
    unsigned short* qb     = (unsigned short*)big;                               // [4][8][P][32]
    float*          scores = (float*)(big + (size_t)BATCH * KCH * PIX * 2);      // [4][19][P]
    unsigned short* attnb  = (unsigned short*)((char*)scores + (size_t)BATCH * NCLS * PIX * 4);
    unsigned short* aggb   = (unsigned short*)((char*)attnb + (size_t)BATCH * PIX * 32 * 2);
    float*          probs  = (float*)((char*)aggb + (size_t)BATCH * VCH * PIX * 2);

    // ---- weight packing + input conversion ----
    {
        int total = NTAPS * 32 * NCH * 256;
        k_pack_conv<<<(total + 255) / 256, 256, 0, stream>>>(w3, convW, total);
    }
    {
        long total = (long)BATCH * NCH * PIX * 32;
        k_cvt_feats<<<(unsigned)((total + 255) / 256), 256, 0, stream>>>(feats, fbT, total);
    }
    k_pack_mat<<<64,  256, 0, stream>>>(aux_w, auxpack, NCLS, MIDC, 2, 16, 0);
    k_pack_mat<<<256, 256, 0, stream>>>(q_w,   qpack,   KCH,  MIDC, 16, 16, 0);
    k_pack_mat<<<64,  256, 0, stream>>>(cls_w, clspack, NCLS, MIDC, 2, 16, 0);
    k_pack_mat<<<256, 256, 0, stream>>>(out_w, owpack,  MIDC, VCH,  32, 8, 0);

    // ---- conv3x3 (WMMA, 2x2 blocked) -> xbuf f32 [b][512][P] ----
    k_conv3x3<<<dim3(PIX / 32, 2, BATCH), 256, 0, stream>>>(fbT, convW, xbuf);

    // ---- BN + relu ----
    k_bnstats<<<MIDC, 256, 0, stream>>>(xbuf, gamma, beta, sa, sb);
    k_bnapply<<<(unsigned)((long)BATCH * MIDC * PIX / 256), 256, 0, stream>>>(xbuf, xb, sa, sb);

    // ---- aux head (WMMA) -> d_out aux half ----
    k_gemm1x1<<<dim3(PIX / 32, 1, BATCH), 32, 0, stream>>>(
        xb, auxpack, 0, 16, 2, 1.f, aux_b, nullptr, NCLS, 1, auxout, nullptr, 0);

    // ---- spatial gather ----
    k_zero<<<1, 256, 0, stream>>>(denom, BATCH * NCLS);
    k_softmax_gather<<<dim3(PIX / 256, BATCH), 256, 0, stream>>>(auxout, probs, denom);
    k_context<<<dim3(MIDC, NCLS, BATCH), 256, 0, stream>>>(probs, denom, xbuf, ctx);

    // ---- k/v projections + per-batch B-packing ----
    k_kv<<<dim3(NCLS, BATCH), 256, 0, stream>>>(ctx, k_w, k_b, v_w, v_b, kmat, vmat);
    for (int b = 0; b < BATCH; ++b) {
        k_pack_mat<<<16, 256, 0, stream>>>(kmat + (long)b * NCLS * KCH, kpack + (long)b * 4096,
                                           NCLS, KCH, 2, 8, 0);
        k_pack_mat<<<16, 256, 0, stream>>>(vmat + (long)b * NCLS * VCH, vpack + (long)b * 4096,
                                           VCH, NCLS, 16, 1, 1);
    }

    // ---- q = x @ q_w^T, scaled 1/sqrt(256) (WMMA) -> qb bf16 interleaved ----
    k_gemm1x1<<<dim3(PIX / 32, 1, BATCH), 256, 0, stream>>>(
        xb, qpack, 0, 16, 16, 0.0625f, nullptr, nullptr, KCH, 0, nullptr, qb, 8);

    // ---- attention scores q.kT (WMMA, per-batch B) -> scores f32 ----
    k_gemm1x1<<<dim3(PIX / 32, 1, BATCH), 32, 0, stream>>>(
        qb, kpack, 4096, 8, 2, 1.f, nullptr, nullptr, NCLS, 1, scores, nullptr, 0);

    // ---- softmax over classes, K-padded 19->32, interleaved ----
    k_softmax_attn<<<dim3(PIX / 256, BATCH), 256, 0, stream>>>(scores, attnb);

    // ---- agg = attn @ v (WMMA, per-batch B) -> aggb bf16 interleaved ----
    k_gemm1x1<<<dim3(PIX / 32, 1, BATCH), 256, 0, stream>>>(
        attnb, vpack, 4096, 1, 16, 1.f, nullptr, nullptr, VCH, 0, nullptr, aggb, 8);

    // ---- obj = agg @ out_w^T + x (WMMA, residual) -> xb (reuse, interleaved) ----
    k_gemm1x1<<<dim3(PIX / 32, 2, BATCH), 256, 0, stream>>>(
        aggb, owpack, 0, 8, 32, 1.f, nullptr, xbuf, MIDC, 0, nullptr, xb, 16);

    // ---- cls head (WMMA) -> d_out logits ----
    k_gemm1x1<<<dim3(PIX / 32, 1, BATCH), 32, 0, stream>>>(
        xb, clspack, 0, 16, 2, 1.f, cls_b, nullptr, NCLS, 1, logits, nullptr, 0);
}
